// AttnDecoderRNN_41137196761115
// MI455X (gfx1250) — matrix operand order repaired and checked by reading, not compiled, not run on hardware
//
#include <hip/hip_runtime.h>
#include <hip/hip_bf16.h>

typedef __bf16 bf16;
typedef bf16 v16bf __attribute__((ext_vector_type(16)));
typedef float v8f  __attribute__((ext_vector_type(8)));

#define B_SZ  2048
#define T_OUTC 24
#define T_INC  48
#define D_OUTC 128
#define D_HIDC 256

union FragU { uint4 u[2]; v16bf v; };

// A fragment: 16x32 bf16 (MxK). Lane L holds row m0+(L&15).
// Lanes 0-15: K = {k0..k0+7, k0+16..k0+23}; lanes 16-31: +8.
__device__ __forceinline__ v16bf load_a_frag(const bf16* __restrict__ A, int lda,
                                             int m0, int k0, int lane) {
  const int m = m0 + (lane & 15);
  const int khalf = (lane >> 4) << 3;   // 0 or 8
  const bf16* p = A + (size_t)m * lda + k0 + khalf;
  FragU f;
  f.u[0] = *reinterpret_cast<const uint4*>(p);        // K = k0+khalf .. +7
  f.u[1] = *reinterpret_cast<const uint4*>(p + 16);   // K = k0+16+khalf .. +7
  return f.v;
}

// B fragment: 32x16 bf16 (KxN) with B[k][n] = W[n][k], W row-major [N,K].
// Lane L holds column n0+(L&15); lanes 0-15: K = {k0..k0+7, k0+16..k0+23}; lanes 16-31: +8.
__device__ __forceinline__ v16bf load_b_frag(const bf16* __restrict__ W, int ldw,
                                             int n0, int k0, int lane) {
  const int n = n0 + (lane & 15);
  const int klane = (lane >> 4) << 3;   // 0 or 8
  const bf16* p = W + (size_t)n * ldw + k0 + klane;
  FragU f;
  f.u[0] = *reinterpret_cast<const uint4*>(p);
  f.u[1] = *reinterpret_cast<const uint4*>(p + 16);
  return f.v;
}

__device__ __forceinline__ v8f wmma_bf16(v16bf a, v16bf b, v8f c) {
  return __builtin_amdgcn_wmma_f32_16x16x32_bf16(false, a, false, b,
                                                 (short)0, c, false, false);
}

// Store one 16x16 f32 tile. C/D layout: VGPR r -> m = mt0 + r + 8*(lane>=16),
// n = nt0 + (lane&15). Per-lane n<N guard is legal post-WMMA.
template <int EPI>
__device__ __forceinline__ void store_tile(v8f acc, void* __restrict__ Cout,
                                           const float* __restrict__ bias,
                                           int mt0, int nt0, int N, int ldc, int lane) {
  const int n = nt0 + (lane & 15);
  if (n >= N) return;
  const float bv = bias[n];
  const int mbase = mt0 + ((lane >> 4) << 3);
  if (EPI == 0) {
    float* C = (float*)Cout;
#pragma unroll
    for (int r = 0; r < 8; ++r)
      C[(size_t)(mbase + r) * ldc + n] = acc[r] + bv;
  } else {
    bf16* C = (bf16*)Cout;
#pragma unroll
    for (int r = 0; r < 8; ++r)
      C[(size_t)(mbase + r) * ldc + n] = (bf16)fmaxf(acc[r] + bv, 0.0f);
  }
}

// out[m,n] = epilogue( sum_k A[m,k]*W[n,k] + bias[n] )
// EPI 0: fp32 store; EPI 1: relu -> bf16 store.
// Block = 256 threads = 8 waves as 4(M)x2(N). Each wave computes a 2x2 grid of
// 16x16 WMMA tiles (32x32 output): 8 b128 loads feed 4 WMMAs per K-step.
// Block tile = 128(M) x 64(N). Weight buffers are padded so loads never need
// guards (EXEC must be all-ones across the WMMA loop); stores guard n < N.
template <int EPI>
__global__ __launch_bounds__(256)
void gemm_wmma_kernel(const bf16* __restrict__ A, const bf16* __restrict__ W,
                      const float* __restrict__ bias, void* __restrict__ Cout,
                      int N, int K, int ldc) {
  const int lane = threadIdx.x & 31;
  const int wave = threadIdx.x >> 5;
  const int m0 = blockIdx.x * 128 + (wave >> 1) * 32;  // M tiles: m0, m0+16
  const int n0 = blockIdx.y * 64 + (wave & 1) * 32;    // N tiles: n0, n0+16
  v8f acc00 = {}, acc01 = {}, acc10 = {}, acc11 = {};
  for (int k0 = 0; k0 < K; k0 += 32) {
    if (k0 + 32 < K)  // CDNA5 speculative prefetch of next A K-slice
      __builtin_prefetch(A + (size_t)(m0 + (lane & 15)) * K + k0 + 32, 0, 1);
    v16bf a0 = load_a_frag(A, K, m0, k0, lane);
    v16bf a1 = load_a_frag(A, K, m0 + 16, k0, lane);
    v16bf b0 = load_b_frag(W, K, n0, k0, lane);
    v16bf b1 = load_b_frag(W, K, n0 + 16, k0, lane);
    acc00 = wmma_bf16(a0, b0, acc00);
    acc01 = wmma_bf16(a0, b1, acc01);
    acc10 = wmma_bf16(a1, b0, acc10);
    acc11 = wmma_bf16(a1, b1, acc11);
  }
  store_tile<EPI>(acc00, Cout, bias, m0,      n0,      N, ldc, lane);
  store_tile<EPI>(acc01, Cout, bias, m0,      n0 + 16, N, ldc, lane);
  store_tile<EPI>(acc10, Cout, bias, m0 + 16, n0,      N, ldc, lane);
  store_tile<EPI>(acc11, Cout, bias, m0 + 16, n0 + 16, N, ldc, lane);
}

// f32 -> bf16 with zero padding (pad rows keep WMMA inputs finite & guard-free).
__global__ void cvt_bf16_pad_kernel(const float* __restrict__ src, bf16* __restrict__ dst,
                                    int nsrc, int ndst) {
  int i = blockIdx.x * 256 + threadIdx.x;
  if (i < ndst) dst[i] = (i < nsrc) ? (bf16)src[i] : (bf16)0.0f;
}

// Copy hidden [2,B,H] fp32 into live-state region (inside d_out) + bf16 shadow.
__global__ void init_state_kernel(const float* __restrict__ hidden,
                                  float* __restrict__ hf, bf16* __restrict__ hbf, int n) {
  int i = blockIdx.x * 256 + threadIdx.x;
  if (i < n) { float v = hidden[i]; hf[i] = v; hbf[i] = (bf16)v; }
}

// attn_in[b, 0:128] = bf16(target[b,t,:]);  attn_in[b, 128:384] = bf16(h1[b,:])
__global__ void attn_prep_kernel(const float* __restrict__ target,
                                 const float* __restrict__ h1,
                                 bf16* __restrict__ attn_in, int t) {
  int idx = blockIdx.x * 256 + threadIdx.x;   // over B*384
  int b = idx / 384, c = idx - b * 384;
  float v = (c < D_OUTC)
              ? target[(size_t)b * T_OUTC * D_OUTC + (size_t)t * D_OUTC + c]
              : h1[(size_t)b * D_HIDC + (c - D_OUTC)];
  attn_in[idx] = (bf16)v;
}

// One block (256 thr) per batch row: softmax(48 logits) -> aw (to d_out),
// ctx[j] = sum_s aw[s]*enc[b,s,j], build comb_in = [bf16(x_t), bf16(ctx)].
__global__ __launch_bounds__(256)
void softmax_ctx_kernel(const float* __restrict__ logits,
                        const float* __restrict__ enc,
                        const float* __restrict__ target,
                        float* __restrict__ aw_out,
                        bf16* __restrict__ comb_in, int t) {
  __shared__ float sl[T_INC];
  __shared__ float sred;
  const int b = blockIdx.x;
  const int tid = threadIdx.x;
  if (tid < T_INC) sl[tid] = logits[(size_t)b * T_INC + tid];
  __syncthreads();
  if (tid == 0) {
    float mx = sl[0];
    for (int i = 1; i < T_INC; ++i) mx = fmaxf(mx, sl[i]);
    sred = mx;
  }
  __syncthreads();
  if (tid < T_INC) sl[tid] = __expf(sl[tid] - sred);
  __syncthreads();
  if (tid == 0) {
    float s = 0.f;
    for (int i = 0; i < T_INC; ++i) s += sl[i];
    sred = 1.0f / s;
  }
  __syncthreads();
  if (tid < T_INC) {
    float a = sl[tid] * sred;
    sl[tid] = a;
    aw_out[(size_t)b * T_OUTC * T_INC + (size_t)t * T_INC + tid] = a;
  }
  __syncthreads();
  float c = 0.f;
  const float* eb = enc + (size_t)b * T_INC * D_HIDC + tid;  // coalesced over tid
#pragma unroll 8
  for (int s = 0; s < T_INC; ++s) c += sl[s] * eb[(size_t)s * D_HIDC];
  bf16* row = comb_in + (size_t)b * 384;
  if (tid < D_OUTC)
    row[tid] = (bf16)target[(size_t)b * T_OUTC * D_OUTC + (size_t)t * D_OUTC + tid];
  row[D_OUTC + tid] = (bf16)c;
}

// PyTorch GRU gate combine (gates r,z,n). gi/gh already include biases.
__global__ void gru_combine_kernel(const float* __restrict__ gi,
                                   const float* __restrict__ gh,
                                   const float* __restrict__ hprev,
                                   float* __restrict__ hnew,
                                   bf16* __restrict__ hnew_bf) {
  int idx = blockIdx.x * 256 + threadIdx.x;   // over B*256
  int b = idx >> 8, j = idx & 255;
  size_t base = (size_t)b * 768;
  float ir = gi[base + j], iz = gi[base + 256 + j], in = gi[base + 512 + j];
  float hr = gh[base + j], hz = gh[base + 256 + j], hn = gh[base + 512 + j];
  float r = 1.f / (1.f + __expf(-(ir + hr)));
  float z = 1.f / (1.f + __expf(-(iz + hz)));
  float nn = tanhf(in + r * hn);
  float h = hprev[idx];
  float out = (1.f - z) * nn + z * h;
  hnew[idx] = out;
  hnew_bf[idx] = (bf16)out;
}

extern "C" void kernel_launch(void* const* d_in, const int* in_sizes, int n_in,
                              void* d_out, int out_size, void* d_ws, size_t ws_size,
                              hipStream_t stream) {
  (void)in_sizes; (void)n_in; (void)out_size; (void)ws_size;
  const float* target = (const float*)d_in[0];
  const float* hidden = (const float*)d_in[1];
  const float* enc    = (const float*)d_in[2];
  const float* W_attn = (const float*)d_in[3];
  const float* b_attn = (const float*)d_in[4];
  const float* W_comb = (const float*)d_in[5];
  const float* b_comb = (const float*)d_in[6];
  const float* W_ih0  = (const float*)d_in[7];
  const float* b_ih0  = (const float*)d_in[8];
  const float* W_hh0  = (const float*)d_in[9];
  const float* b_hh0  = (const float*)d_in[10];
  const float* W_ih1  = (const float*)d_in[11];
  const float* b_ih1  = (const float*)d_in[12];
  const float* W_hh1  = (const float*)d_in[13];
  const float* b_hh1  = (const float*)d_in[14];
  const float* W1 = (const float*)d_in[15];
  const float* b1 = (const float*)d_in[16];
  const float* W2 = (const float*)d_in[17];
  const float* b2 = (const float*)d_in[18];
  const float* W3 = (const float*)d_in[19];
  const float* b3 = (const float*)d_in[20];

  // d_out layout: output [B,24,128] | h_final [2,B,256] | attn_weights [B,24,48]
  float* out_y  = (float*)d_out;
  float* hf     = out_y + (size_t)B_SZ * T_OUTC * D_OUTC;
  float* out_aw = hf + (size_t)2 * B_SZ * D_HIDC;
  float* h0 = hf;
  float* h1 = hf + (size_t)B_SZ * D_HIDC;

  // workspace carve-out (256B aligned)
  char* wsp = (char*)d_ws;
  auto carve = [&](size_t bytes) {
    char* p = wsp; wsp += (bytes + 255) & ~(size_t)255; return p;
  };
  bf16* Wattn_bf = (bf16*)carve((size_t)64 * 384 * 2);   // padded 48 -> 64 rows
  bf16* Wcomb_bf = (bf16*)carve((size_t)128 * 384 * 2);
  bf16* Wih0_bf  = (bf16*)carve((size_t)768 * 128 * 2);
  bf16* Whh0_bf  = (bf16*)carve((size_t)768 * 256 * 2);
  bf16* Wih1_bf  = (bf16*)carve((size_t)768 * 256 * 2);
  bf16* Whh1_bf  = (bf16*)carve((size_t)768 * 256 * 2);
  bf16* W1_bf    = (bf16*)carve((size_t)256 * 256 * 2);
  bf16* W2_bf    = (bf16*)carve((size_t)128 * 256 * 2);
  bf16* W3_bf    = (bf16*)carve((size_t)128 * 128 * 2);
  bf16* attn_in  = (bf16*)carve((size_t)B_SZ * 384 * 2);
  bf16* comb_in  = (bf16*)carve((size_t)B_SZ * 384 * 2);
  bf16* x_bf     = (bf16*)carve((size_t)B_SZ * 128 * 2);
  bf16* h01_bf   = (bf16*)carve((size_t)2 * B_SZ * 256 * 2);
  bf16* h0_bf = h01_bf;
  bf16* h1_bf = h01_bf + (size_t)B_SZ * 256;
  bf16* o1_bf    = (bf16*)carve((size_t)B_SZ * 256 * 2);
  bf16* o2_bf    = (bf16*)carve((size_t)B_SZ * 128 * 2);
  float* logits  = (float*)carve((size_t)B_SZ * 48 * 4);
  float* gi      = (float*)carve((size_t)B_SZ * 768 * 4);
  float* gh      = (float*)carve((size_t)B_SZ * 768 * 4);

  // one-time per call: weights -> bf16 (L2-resident thereafter), init state
  struct CvtJob { const float* s; bf16* d; int nsrc; int ndst; };
  const CvtJob jobs[9] = {
    {W_attn, Wattn_bf, 48 * 384, 64 * 384},   // zero-padded tail rows
    {W_comb, Wcomb_bf, 128 * 384, 128 * 384},
    {W_ih0, Wih0_bf, 768 * 128, 768 * 128},
    {W_hh0, Whh0_bf, 768 * 256, 768 * 256},
    {W_ih1, Wih1_bf, 768 * 256, 768 * 256},
    {W_hh1, Whh1_bf, 768 * 256, 768 * 256},
    {W1, W1_bf, 256 * 256, 256 * 256},
    {W2, W2_bf, 128 * 256, 128 * 256},
    {W3, W3_bf, 128 * 128, 128 * 128}};
  for (int i = 0; i < 9; ++i)
    cvt_bf16_pad_kernel<<<(jobs[i].ndst + 255) / 256, 256, 0, stream>>>(
        jobs[i].s, jobs[i].d, jobs[i].nsrc, jobs[i].ndst);
  init_state_kernel<<<(2 * B_SZ * 256 + 255) / 256, 256, 0, stream>>>(
      hidden, hf, h01_bf, 2 * B_SZ * 256);

  const dim3 blk(256);
  const int GX = B_SZ / 128;   // 16 blocks over M
  for (int t = 0; t < T_OUTC; ++t) {
    attn_prep_kernel<<<(B_SZ * 384) / 256, blk, 0, stream>>>(target, h1, attn_in, t);
    // logits = attn_in @ W_attn^T + b_attn        [2048 x 48], K=384
    gemm_wmma_kernel<0><<<dim3(GX, 1), blk, 0, stream>>>(attn_in, Wattn_bf, b_attn,
                                                         logits, 48, 384, 48);
    softmax_ctx_kernel<<<B_SZ, blk, 0, stream>>>(logits, enc, target, out_aw, comb_in, t);
    // x = relu(comb_in @ W_comb^T + b_comb)       [2048 x 128], K=384
    gemm_wmma_kernel<1><<<dim3(GX, 2), blk, 0, stream>>>(comb_in, Wcomb_bf, b_comb,
                                                         x_bf, 128, 384, 128);
    // GRU layer 0
    gemm_wmma_kernel<0><<<dim3(GX, 12), blk, 0, stream>>>(x_bf, Wih0_bf, b_ih0,
                                                          gi, 768, 128, 768);
    gemm_wmma_kernel<0><<<dim3(GX, 12), blk, 0, stream>>>(h0_bf, Whh0_bf, b_hh0,
                                                          gh, 768, 256, 768);
    gru_combine_kernel<<<(B_SZ * 256) / 256, blk, 0, stream>>>(gi, gh, h0, h0, h0_bf);
    // GRU layer 1
    gemm_wmma_kernel<0><<<dim3(GX, 12), blk, 0, stream>>>(h0_bf, Wih1_bf, b_ih1,
                                                          gi, 768, 256, 768);
    gemm_wmma_kernel<0><<<dim3(GX, 12), blk, 0, stream>>>(h1_bf, Whh1_bf, b_hh1,
                                                          gh, 768, 256, 768);
    gru_combine_kernel<<<(B_SZ * 256) / 256, blk, 0, stream>>>(gi, gh, h1, h1, h1_bf);
    // output MLP
    gemm_wmma_kernel<1><<<dim3(GX, 4), blk, 0, stream>>>(h1_bf, W1_bf, b1,
                                                         o1_bf, 256, 256, 256);
    gemm_wmma_kernel<1><<<dim3(GX, 2), blk, 0, stream>>>(o1_bf, W2_bf, b2,
                                                         o2_bf, 128, 256, 128);
    // out[b,t,:] = o2 @ W3^T + b3  (strided fp32 store into output tensor)
    gemm_wmma_kernel<0><<<dim3(GX, 2), blk, 0, stream>>>(o2_bf, W3_bf, b3,
                                                         out_y + (size_t)t * D_OUTC,
                                                         128, 128, T_OUTC * D_OUTC);
  }
}
